// Expansion_44057774522742
// MI455X (gfx1250) — matrix operand order repaired and checked by reading, not compiled, not run on hardware
//
#include <hip/hip_runtime.h>
#include <hip/hip_bf16.h>

// ---------------------------------------------------------------------------
// Types for CDNA5 WMMA
// ---------------------------------------------------------------------------
typedef __attribute__((ext_vector_type(16))) _Float16 v16h;
typedef __attribute__((ext_vector_type(8)))  float    v8f;

#define WCS 132   // padded LDS stride for the 16x128 weight chunk (bank-conflict pad)
#define PW  17536 // NUM_PATH_WEIGHT
#define NB  336   // NUM_BIAS

// ---------------------------------------------------------------------------
// Device Wigner-3j generation (Racah formula + complex->real SH basis change)
// One tiny kernel, double precision, 363 coefficients total.
// ---------------------------------------------------------------------------
__device__ static double dfact(int n) {
    double r = 1.0;
    for (int i = 2; i <= n; ++i) r *= (double)i;
    return r;
}

__device__ static void build_u(int l, double* Ur, double* Ui) {
    int n = 2 * l + 1;
    for (int a = 0; a < n * n; ++a) { Ur[a] = 0.0; Ui[a] = 0.0; }
    Ur[l * n + l] = 1.0;
    double s = 1.0 / sqrt(2.0);
    for (int m = 1; m <= l; ++m) {
        double sg = (m & 1) ? -1.0 : 1.0;
        Ur[(l + m) * n + (l + m)] = sg * s;
        Ur[(l + m) * n + (l - m)] = s;
        Ui[(l - m) * n + (l - m)] = s;
        Ui[(l - m) * n + (l + m)] = -sg * s;
    }
}

__global__ void w3j_kernel(float* __restrict__ outW) {
    int p = threadIdx.x;
    if (p >= 11) return;
    const int L1a[11]  = {0, 1, 2, 0, 1, 1, 2, 0, 1, 2, 2};
    const int L2a[11]  = {0, 1, 2, 1, 0, 2, 1, 2, 1, 0, 2};
    const int L3a[11]  = {0, 0, 0, 1, 1, 1, 1, 2, 2, 2, 2};
    const int OFFa[11] = {0, 1, 10, 35, 44, 53, 98, 143, 168, 213, 238};
    int l1 = L1a[p], l2 = L2a[p], l3 = L3a[p];
    int n1 = 2 * l1 + 1, n2 = 2 * l2 + 1, n3 = 2 * l3 + 1;

    double W[125];
    for (int a = 0; a < 125; ++a) W[a] = 0.0;
    double tri = dfact(l1 + l2 - l3) * dfact(l1 - l2 + l3) * dfact(l2 + l3 - l1)
               / dfact(l1 + l2 + l3 + 1);
    for (int m1 = -l1; m1 <= l1; ++m1) {
        for (int m2 = -l2; m2 <= l2; ++m2) {
            int m3 = -m1 - m2;
            if (m3 < -l3 || m3 > l3) continue;
            double pref = sqrt(tri * dfact(l1 + m1) * dfact(l1 - m1) * dfact(l2 + m2)
                               * dfact(l2 - m2) * dfact(l3 + m3) * dfact(l3 - m3));
            int t0 = 0;
            if (l2 - l3 - m1 > t0) t0 = l2 - l3 - m1;
            if (l1 - l3 + m2 > t0) t0 = l1 - l3 + m2;
            int t1 = l1 + l2 - l3;
            if (l1 - m1 < t1) t1 = l1 - m1;
            if (l2 + m2 < t1) t1 = l2 + m2;
            double s = 0.0;
            for (int t = t0; t <= t1; ++t) {
                double d = dfact(t) * dfact(l1 + l2 - l3 - t) * dfact(l1 - m1 - t)
                         * dfact(l2 + m2 - t) * dfact(l3 - l2 + m1 + t) * dfact(l3 - l1 - m2 + t);
                s += ((t & 1) ? -1.0 : 1.0) / d;
            }
            int e = l1 - l2 - m3;
            double ph = (((e % 2) + 2) % 2) ? -1.0 : 1.0;
            W[(m1 + l1) * n2 * n3 + (m2 + l2) * n3 + (m3 + l3)] = ph * pref * s;
        }
    }
    double U1r[25], U1i[25], U2r[25], U2i[25], U3r[25], U3i[25];
    build_u(l1, U1r, U1i);
    build_u(l2, U2r, U2i);
    build_u(l3, U3r, U3i);
    for (int i = 0; i < n1; ++i)
        for (int j = 0; j < n2; ++j)
            for (int k = 0; k < n3; ++k) {
                double accr = 0.0;
                for (int m = 0; m < n1; ++m) {
                    double ar = U1r[i * n1 + m], ai = U1i[i * n1 + m];
                    if (ar == 0.0 && ai == 0.0) continue;
                    for (int n = 0; n < n2; ++n) {
                        double br = U2r[j * n2 + n], bi = U2i[j * n2 + n];
                        if (br == 0.0 && bi == 0.0) continue;
                        double pr = ar * br - ai * bi;
                        double pi = ar * bi + ai * br;
                        for (int q = 0; q < n3; ++q) {
                            double w = W[m * n2 * n3 + n * n3 + q];
                            if (w == 0.0) continue;
                            double cr = U3r[k * n3 + q], ci = U3i[k * n3 + q];
                            accr += (pr * cr - pi * ci) * w;
                        }
                    }
                }
                outW[OFFa[p] + (i * n2 + j) * n3 + k] = (float)accr;
            }
}

// ---------------------------------------------------------------------------
// Prep: per-sample 2-layer MLPs. H (weight branch hidden) -> f16 for WMMA A.
// Bias branch fully evaluated to f32 [N][336].
// ---------------------------------------------------------------------------
__global__ __launch_bounds__(64) void prep_kernel(
    const float* __restrict__ ne,
    const float* __restrict__ w1w, const float* __restrict__ b1w,
    const float* __restrict__ w1b, const float* __restrict__ b1b,
    const float* __restrict__ w2b, const float* __restrict__ b2b,
    _Float16* __restrict__ Hout, float* __restrict__ biasOut)
{
    __shared__ float neS[128];
    __shared__ float hbS[64];
    int b = blockIdx.x, t = threadIdx.x;
    neS[t]      = ne[b * 128 + t];
    neS[t + 64] = ne[b * 128 + 64 + t];
    __syncthreads();
    float aw = b1w[t], ab = b1b[t];
    #pragma unroll 4
    for (int c = 0; c < 128; ++c) {
        float x = neS[c];
        aw += x * w1w[c * 64 + t];
        ab += x * w1b[c * 64 + t];
    }
    float hw = aw / (1.0f + expf(-aw));
    float hb = ab / (1.0f + expf(-ab));
    Hout[b * 64 + t] = (_Float16)hw;
    hbS[t] = hb;
    __syncthreads();
    for (int p = t; p < NB; p += 64) {
        float acc = b2b[p];
        #pragma unroll 8
        for (int c = 0; c < 64; ++c) acc += hbS[c] * w2b[c * NB + p];
        biasOut[b * NB + p] = acc;
    }
}

// Transpose + convert w2w [64][17536] f32 -> w2wT [17536][64] f16 (B-fragment friendly)
__global__ __launch_bounds__(256) void tr_kernel(const float* __restrict__ w2w,
                                                 _Float16* __restrict__ w2wT)
{
    int idx = blockIdx.x * 256 + threadIdx.x;   // 17536*64 total
    int p = idx >> 6;
    int c = idx & 63;
    w2wT[idx] = (_Float16)w2w[c * PW + p];
}

// ---------------------------------------------------------------------------
// Fused expansion: per path, WMMA weight chunks (16 samples x 128 cols) into
// LDS, immediate Sum_w contraction + bias + w3j into registers.
// ---------------------------------------------------------------------------
template<int MULIN, int LI, int WOFF, int BOFF, int XOFF, int TJOFF,
         int MUL1, int MUL2, int NSLOT, int D1, int D2>
__device__ __forceinline__ void run_path(
    int tid, int sample0, v16h a0, v16h a1,
    const _Float16* __restrict__ w2wT, const float* __restrict__ b2w,
    const float* __restrict__ biasv, const float* __restrict__ w3j,
    const float (* __restrict__ Xs)[120], float* __restrict__ Wc,
    float (&outr)[NSLOT][D1][D2])
{
    constexpr int UV  = MUL1 * MUL2;
    constexpr int KD  = 2 * LI + 1;
    constexpr int NCH = (MULIN * UV) / 128;
    int lane    = tid & 31;
    int waveid  = tid >> 5;
    int colLane = lane & 15;
    int halfSel = lane >> 4;

    float res[NSLOT][KD];
    #pragma unroll
    for (int s = 0; s < NSLOT; ++s)
        #pragma unroll
        for (int k = 0; k < KD; ++k) res[s][k] = 0.0f;

    for (int ch = 0; ch < NCH; ++ch) {
        int rel0 = ch * 128;
        #pragma unroll
        for (int tt = 0; tt < 2; ++tt) {
            int t    = waveid * 2 + tt;
            int colG = WOFF + rel0 + t * 16 + colLane;
            const _Float16* bp = w2wT + colG * 64 + halfSel * 16;
            v16h bf0 = *(const v16h*)(bp);
            v16h bf1 = *(const v16h*)(bp + 32);
            if (ch + 1 < NCH) __builtin_prefetch(bp + 128 * 64, 0, 1);
            float bv = b2w[colG];
            v8f c;
            #pragma unroll
            for (int g = 0; g < 8; ++g) c[g] = bv;
            c = __builtin_amdgcn_wmma_f32_16x16x32_f16(false, a0, false, bf0,
                                                       (short)0, c, false, false);
            c = __builtin_amdgcn_wmma_f32_16x16x32_f16(false, a1, false, bf1,
                                                       (short)0, c, false, false);
            int colS = t * 16 + colLane;
            #pragma unroll
            for (int g = 0; g < 8; ++g)
                Wc[(halfSel * 8 + g) * WCS + colS] = c[g];
        }
        __syncthreads();
        #pragma unroll
        for (int s = 0; s < NSLOT; ++s) {
            int gs = s * 128 + tid;
            int b  = gs / UV;
            int uv = gs % UV;
            int dlo = rel0 - uv;
            int wlo = dlo > 0 ? (dlo + UV - 1) / UV : 0;
            int dhi = rel0 + 127 - uv;
            int whi = dhi < 0 ? -1 : dhi / UV;
            if (whi > MULIN - 1) whi = MULIN - 1;
            const float* xb = Xs[b] + XOFF;
            for (int w = wlo; w <= whi; ++w) {
                float wv = Wc[b * WCS + (w * UV + uv - rel0)];
                #pragma unroll
                for (int k = 0; k < KD; ++k)
                    res[s][k] += wv * xb[w * KD + k];
            }
        }
        __syncthreads();
    }
    if constexpr (BOFF >= 0) {
        #pragma unroll
        for (int s = 0; s < NSLOT; ++s) {
            int gs = s * 128 + tid;
            int b  = gs / UV;
            int uv = gs % UV;
            res[s][0] += biasv[(sample0 + b) * NB + BOFF + uv];
        }
    }
    constexpr float inv = 1.0f / (float)MULIN;
    #pragma unroll
    for (int s = 0; s < NSLOT; ++s)
        #pragma unroll
        for (int i = 0; i < D1; ++i)
            #pragma unroll
            for (int j = 0; j < D2; ++j) {
                float acc = 0.0f;
                #pragma unroll
                for (int k = 0; k < KD; ++k)
                    acc += w3j[TJOFF + (i * D2 + j) * KD + k] * res[s][k];
                outr[s][i][j] += acc * inv;
            }
}

template<int MUL1, int MUL2, int L1, int L2, int ROW0, int COL0,
         int MI0, int LI0, int WO0, int BO0, int XO0, int TJ0,
         bool HAS2, int MI1, int LI1, int WO1, int XO1, int TJ1>
__device__ __forceinline__ void run_jk(
    int tid, int sample0, v16h a0, v16h a1,
    const _Float16* __restrict__ w2wT, const float* __restrict__ b2w,
    const float* __restrict__ biasv, const float* __restrict__ w3j,
    const float (* __restrict__ Xs)[120], float* __restrict__ Wc,
    float* __restrict__ out)
{
    constexpr int UV    = MUL1 * MUL2;
    constexpr int NSLOT = (16 * UV) / 128;
    constexpr int D1    = 2 * L1 + 1;
    constexpr int D2    = 2 * L2 + 1;
    float outr[NSLOT][D1][D2];
    #pragma unroll
    for (int s = 0; s < NSLOT; ++s)
        #pragma unroll
        for (int i = 0; i < D1; ++i)
            #pragma unroll
            for (int j = 0; j < D2; ++j) outr[s][i][j] = 0.0f;

    run_path<MI0, LI0, WO0, BO0, XO0, TJ0, MUL1, MUL2, NSLOT, D1, D2>(
        tid, sample0, a0, a1, w2wT, b2w, biasv, w3j, Xs, Wc, outr);
    if constexpr (HAS2)
        run_path<MI1, LI1, WO1, -1, XO1, TJ1, MUL1, MUL2, NSLOT, D1, D2>(
            tid, sample0, a0, a1, w2wT, b2w, biasv, w3j, Xs, Wc, outr);

    #pragma unroll
    for (int s = 0; s < NSLOT; ++s) {
        int gs = s * 128 + tid;
        int b  = gs / UV;
        int uv = gs % UV;
        int u  = uv / MUL2;
        int v  = uv % MUL2;
        float* ob = out + (size_t)(sample0 + b) * 3600;
        #pragma unroll
        for (int i = 0; i < D1; ++i)
            #pragma unroll
            for (int j = 0; j < D2; ++j)
                ob[(ROW0 + u * D1 + i) * 60 + (COL0 + v * D2 + j)] = outr[s][i][j];
    }
}

__global__ __launch_bounds__(128) void expansion_kernel(
    const _Float16* __restrict__ H, const _Float16* __restrict__ w2wT,
    const float* __restrict__ b2w, const float* __restrict__ biasv,
    const float* __restrict__ x_in, const float* __restrict__ w3j,
    float* __restrict__ out)
{
    __shared__ _Float16 Hs[16 * 64];
    __shared__ float    Xs[16][120];
    __shared__ float    Wc[16 * WCS];
    int tid     = threadIdx.x;
    int sample0 = blockIdx.x * 16;

    const _Float16* hsrc = H + sample0 * 64;
    #pragma unroll
    for (int r = 0; r < 8; ++r) Hs[tid + r * 128] = hsrc[tid + r * 128];
    const float* xsrc = x_in + sample0 * 120;
    #pragma unroll
    for (int r = 0; r < 15; ++r) ((float*)Xs)[tid + r * 128] = xsrc[tid + r * 128];
    __syncthreads();

    // A fragments (16x32 per fragment, contiguous K order matching B):
    // lane<16 holds row=lane, K {0..15}; lane>=16 holds
    // row=lane-16, K {16..31}.
    int lane = tid & 31;
    int row  = lane & 15;
    int kb   = (lane >> 4) * 16;
    v16h a0, a1;
    #pragma unroll
    for (int e = 0; e < 16; ++e) {
        int k = kb + e;
        a0[e] = Hs[row * 64 + k];
        a1[e] = Hs[row * 64 + 32 + k];
    }

    #define RUNARGS tid, sample0, a0, a1, w2wT, b2w, biasv, w3j, \
                    (const float(*)[120])Xs, Wc, out
    switch (blockIdx.y) {
    case 0: run_jk<16,16,0,0,  0, 0,  32,0,    0,  0, 0,  0, false,1,0,0,0,0>(RUNARGS); break;
    case 1: run_jk<16, 8,0,1,  0,16,  16,1,10752, -1,32, 35, false,1,0,0,0,0>(RUNARGS); break;
    case 2: run_jk<16, 4,0,2,  0,40,   8,2,15872, -1,80,143, false,1,0,0,0,0>(RUNARGS); break;
    case 3: run_jk< 8,16,1,0, 16, 0,  16,1,12800, -1,32, 44, false,1,0,0,0,0>(RUNARGS); break;
    case 4: run_jk< 8, 8,1,1, 16,16,  32,0, 8192,256, 0,  1, true, 8,2,16384,80,168>(RUNARGS); break;
    case 5: run_jk< 8, 4,1,2, 16,40,  16,1,14848, -1,32, 53, false,1,0,0,0,0>(RUNARGS); break;
    case 6: run_jk< 4,16,2,0, 40, 0,   8,2,16896, -1,80,213, false,1,0,0,0,0>(RUNARGS); break;
    case 7: run_jk< 4, 8,2,1, 40,16,  16,1,15360, -1,32, 98, false,1,0,0,0,0>(RUNARGS); break;
    case 8: run_jk< 4, 4,2,2, 40,40,  32,0,10240,320, 0, 10, true, 8,2,17408,80,238>(RUNARGS); break;
    }
    #undef RUNARGS
}

// ---------------------------------------------------------------------------
// Launch
// ---------------------------------------------------------------------------
extern "C" void kernel_launch(void* const* d_in, const int* in_sizes, int n_in,
                              void* d_out, int out_size, void* d_ws, size_t ws_size,
                              hipStream_t stream)
{
    const float* x_in = (const float*)d_in[0];
    const float* ne   = (const float*)d_in[1];
    const float* w1w  = (const float*)d_in[2];
    const float* b1w  = (const float*)d_in[3];
    const float* w2w  = (const float*)d_in[4];
    const float* b2w  = (const float*)d_in[5];
    const float* w1b  = (const float*)d_in[6];
    const float* b1b  = (const float*)d_in[7];
    const float* w2b  = (const float*)d_in[8];
    const float* b2b  = (const float*)d_in[9];
    float* out = (float*)d_out;

    int N = in_sizes[0] / 120;   // 8192

    char* ws = (char*)d_ws;
    _Float16* w2wT = (_Float16*)(ws);                 // PW*64*2 = 2,244,608 B
    float*    w3jw = (float*)(ws + 2244608);          // 363*4  = 1,452 B
    _Float16* Hws  = (_Float16*)(ws + 2246144);       // N*64*2
    float*    bws  = (float*)(ws + 2246144 + (size_t)N * 128);  // N*336*4

    w3j_kernel<<<dim3(1), dim3(32), 0, stream>>>(w3jw);
    prep_kernel<<<dim3(N), dim3(64), 0, stream>>>(ne, w1w, b1w, w1b, b1b, w2b, b2b,
                                                  Hws, bws);
    tr_kernel<<<dim3((PW * 64) / 256), dim3(256), 0, stream>>>(w2w, w2wT);
    expansion_kernel<<<dim3(N / 16, 9), dim3(128), 0, stream>>>(
        Hws, w2wT, b2w, bws, x_in, w3jw, out);
}
